// BinaryDense_39573828666264
// MI455X (gfx1250) — 8-bit matrix operand order repaired and checked by reading, compiled, not run on hardware
//
#include <hip/hip_runtime.h>

typedef __attribute__((ext_vector_type(8))) int   v8i;
typedef __attribute__((ext_vector_type(4))) float v4f;

#define D_IN   4096
#define N_ROWS 8192
#define N_COLS 4096
#define KTILES (D_IN / 64)
#define TILE_BYTES 1024

__device__ __forceinline__ int sgn8(float v) {
    return (v > 0.0f) ? 1 : ((v < 0.0f) ? -1 : 0);
}

__global__ __launch_bounds__(256) void pack_x_kernel(const float* __restrict__ x,
                                                     char* __restrict__ apk) {
    int tid = blockIdx.x * 256 + threadIdx.x;
    int m  = tid >> 10;
    int k  = (tid & 1023) << 2;
    v4f v = *(const v4f*)(x + (long)m * D_IN + k);
    int packed = (sgn8(v.x) & 0xFF)
               | ((sgn8(v.y) & 0xFF) << 8)
               | ((sgn8(v.z) & 0xFF) << 16)
               | ((sgn8(v.w) & 0xFF) << 24);
    int mt = m >> 4, M = m & 15;
    int kt = k >> 6, K = k & 63;
    int lane = M + 16 * ((K & 31) >> 4);
    int V    = 4 * (K >> 5) + ((K & 15) >> 2);
    long base = (long)(mt * KTILES + kt) * TILE_BYTES;
    *(int*)(apk + base + lane * 32 + V * 4) = packed;
}

__global__ __launch_bounds__(256) void pack_w_kernel(const float* __restrict__ w,
                                                     char* __restrict__ bpk) {
    int tid = blockIdx.x * 256 + threadIdx.x;
    int k = tid >> 10;
    int n = (tid & 1023) << 2;
    v4f v = *(const v4f*)(w + (long)k * N_COLS + n);
    int s[4] = { sgn8(v.x), sgn8(v.y), sgn8(v.z), sgn8(v.w) };
    int nt = n >> 4;
    int kt = k >> 6, K = k & 63;
    int V  = 4 * (K >> 5) + ((K & 15) >> 2);
    int byt = K & 3;
    long base = (long)(nt * KTILES + kt) * TILE_BYTES;
    int n0 = n & 15;
    #pragma unroll
    for (int j = 0; j < 4; ++j) {
        int lane = (n0 + j) + 16 * ((K & 31) >> 4);
        bpk[base + lane * 32 + V * 4 + byt] = (char)s[j];
    }
}

__global__ __launch_bounds__(256) void bgemm_kernel(const char* __restrict__ apk,
                                                    const char* __restrict__ bpk,
                                                    float* __restrict__ out) {
    const int lane = threadIdx.x & 31;
    const int wave = threadIdx.x >> 5;
    const int wm = wave >> 2;
    const int wn = wave & 3;
    const int mBase = blockIdx.y * 128 + wm * 64;
    const int nBase = blockIdx.x * 256 + wn * 64;

    const char* aptr = apk + (long)(mBase >> 4) * KTILES * TILE_BYTES + lane * 32;
    const char* bptr = bpk + (long)(nBase >> 4) * KTILES * TILE_BYTES + lane * 32;
    const long tileStride = (long)KTILES * TILE_BYTES;

    v8i acc[4][4];
    #pragma unroll
    for (int i = 0; i < 4; ++i)
        #pragma unroll
        for (int j = 0; j < 4; ++j)
            acc[i][j] = (v8i)(0);

    for (int kt = 0; kt < KTILES; ++kt) {
        const long ko = (long)kt * TILE_BYTES;
        v8i a[4], b[4];
        #pragma unroll
        for (int i = 0; i < 4; ++i)
            a[i] = *(const v8i*)(aptr + (long)i * tileStride + ko);
        #pragma unroll
        for (int j = 0; j < 4; ++j)
            b[j] = *(const v8i*)(bptr + (long)j * tileStride + ko);

        #pragma unroll
        for (int i = 0; i < 4; ++i) {
            #pragma unroll
            for (int j = 0; j < 4; ++j) {
                acc[i][j] = __builtin_amdgcn_wmma_i32_16x16x64_iu8(
                    true, a[i], true, b[j], acc[i][j], false, false);
            }
        }
    }

    const int halfsel = lane >> 4;
    const int col0 = lane & 15;
    #pragma unroll
    for (int i = 0; i < 4; ++i) {
        #pragma unroll
        for (int j = 0; j < 4; ++j) {
            const int rowb = mBase + i * 16 + halfsel * 8;
            const int col  = nBase + j * 16 + col0;
            #pragma unroll
            for (int r = 0; r < 8; ++r) {
                out[(long)(rowb + r) * N_COLS + col] = (float)acc[i][j][r];
            }
        }
    }
}

extern "C" void kernel_launch(void* const* d_in, const int* in_sizes, int n_in,
                              void* d_out, int out_size, void* d_ws, size_t ws_size,
                              hipStream_t stream) {
    const float* x = (const float*)d_in[0];
    const float* w = (const float*)d_in[1];
    float* out = (float*)d_out;

    char* apk = (char*)d_ws;
    char* bpk = (char*)d_ws + (size_t)N_ROWS * D_IN;

    pack_x_kernel<<<(N_ROWS * (D_IN / 4)) / 256, 256, 0, stream>>>(x, apk);
    pack_w_kernel<<<(D_IN * (N_COLS / 4)) / 256, 256, 0, stream>>>(w, bpk);

    dim3 grid(N_COLS / 256, N_ROWS / 128);
    bgemm_kernel<<<grid, 256, 0, stream>>>(apk, bpk, out);
}
